// EncoderBlock_70927089926617
// MI455X (gfx1250) — matrix operand order repaired and checked by reading, not compiled, not run on hardware
//
#include <hip/hip_runtime.h>
#include <math.h>

// ---------------------------------------------------------------------------
// Types for CDNA5 WMMA (wave32): v_wmma_f32_16x16x32_bf16
// ---------------------------------------------------------------------------
typedef __bf16 bf16_t;
typedef __attribute__((ext_vector_type(16))) __bf16 v16bf;
typedef __attribute__((ext_vector_type(8)))  __bf16 v8bf;
typedef __attribute__((ext_vector_type(4)))  __bf16 v4bf;
typedef __attribute__((ext_vector_type(8)))  float  v8f;

static __device__ __forceinline__ v8f wmma_bf16(v16bf a, v16bf b, v8f c) {
  // D = A(16x32 bf16) x B(32x16 bf16) + C(16x16 f32)
  return __builtin_amdgcn_wmma_f32_16x16x32_bf16(
      /*neg_a=*/false, a, /*neg_b=*/false, b,
      /*c_mod=*/(short)0, c, /*reuse_a=*/false, /*reuse_b=*/false);
}

// B fragment: 16 contiguous bf16 (lane covers K = kb..kb+15, element e <-> K=kb+e)
static __device__ __forceinline__ v16bf load_bfrag(const bf16_t* __restrict__ p) {
  v8bf x0 = *(const v8bf*)p;
  v8bf x1 = *(const v8bf*)(p + 8);
  v16bf r;
#pragma unroll
  for (int e = 0; e < 8; ++e) { r[e] = x0[e]; r[e + 8] = x1[e]; }
  return r;
}

// A fragment: 16 contiguous bf16
// (lane l: row = l&15, k0 = (l>>4)*16; element e -> K=k0+e)
static __device__ __forceinline__ v16bf load_afrag(const bf16_t* __restrict__ p) {
  v8bf x0 = *(const v8bf*)p;
  v8bf x1 = *(const v8bf*)(p + 8);
  v16bf r;
#pragma unroll
  for (int e = 0; e < 8; ++e) { r[e] = x0[e]; r[e + 8] = x1[e]; }
  return r;
}

// ---------------------------------------------------------------------------
// Model constants
// ---------------------------------------------------------------------------
#define BATCH 2
#define SEQ   2048
#define TOK   (BATCH * SEQ)   // 4096
#define DMODEL 1024
#define NHEAD 16
#define DKEY  64
#define DFF   4096

// ---------------------------------------------------------------------------
// Weight transpose + fp32 -> bf16 convert: Wt[n*K + k] = W[k*N + n]
// ---------------------------------------------------------------------------
__global__ void transpose_to_bf16_kernel(const float* __restrict__ w,
                                         bf16_t* __restrict__ wt,
                                         int K, int N) {
  size_t idx = (size_t)blockIdx.x * blockDim.x + threadIdx.x;
  size_t total = (size_t)K * N;
  if (idx >= total) return;
  int n = (int)(idx / K);
  int k = (int)(idx % K);
  wt[idx] = (bf16_t)w[(size_t)k * N + n];
}

// ---------------------------------------------------------------------------
// LayerNorm (faithful: unbiased std (ddof=1), eps added to std, scalar a/b)
// One block of 256 threads per row of D=1024; emit bf16 for WMMA consumption.
// ---------------------------------------------------------------------------
__global__ void layernorm_bf16_kernel(const float* __restrict__ x,
                                      bf16_t* __restrict__ out,
                                      const float* __restrict__ alpha,
                                      const float* __restrict__ beta) {
  const int D = DMODEL;
  int row = blockIdx.x;
  int tid = threadIdx.x;                  // 256 threads, 4 floats each
  const float* xr = x + (size_t)row * D;
  float4 v = *(const float4*)(xr + tid * 4);

  __shared__ float red[256];
  red[tid] = v.x + v.y + v.z + v.w;
  __syncthreads();
#pragma unroll
  for (int off = 128; off > 0; off >>= 1) {
    if (tid < off) red[tid] += red[tid + off];
    __syncthreads();
  }
  float mean = red[0] * (1.0f / D);
  __syncthreads();

  float d0 = v.x - mean, d1 = v.y - mean, d2 = v.z - mean, d3 = v.w - mean;
  red[tid] = d0 * d0 + d1 * d1 + d2 * d2 + d3 * d3;
  __syncthreads();
#pragma unroll
  for (int off = 128; off > 0; off >>= 1) {
    if (tid < off) red[tid] += red[tid + off];
    __syncthreads();
  }
  float stdv = sqrtf(red[0] * (1.0f / (D - 1)));   // torch.std default ddof=1
  float inv  = alpha[0] / (stdv + 1e-6f);
  float bb   = beta[0];

  v4bf o;
  o[0] = (bf16_t)(d0 * inv + bb);
  o[1] = (bf16_t)(d1 * inv + bb);
  o[2] = (bf16_t)(d2 * inv + bb);
  o[3] = (bf16_t)(d3 * inv + bb);
  *(v4bf*)(out + (size_t)row * D + tid * 4) = o;
}

// ---------------------------------------------------------------------------
// WMMA GEMM: C[M,N] = A[M,K](bf16) @ Wt[N,K](bf16)^T + bias, with epilogues.
//   mode 0: scale, write bf16 to per-head [B,H,S,Dk] layout       (Q / K proj)
//   mode 1: write bf16 transposed per head [B,H,Dk,S]             (V proj)
//   mode 2: f32 out[m*N+n] = acc + bias + residual[m*N+n]         (O / FFN2)
//   mode 3: bf16 out = relu(acc + bias)                           (FFN1)
// Block: 256 threads = 8 waves; wave w -> rows [bx*128+w*16,+16), cols [by*64,+64)
// K loop is software-pipelined: fragments for step kk+32 are fetched before the
// four WMMAs of step kk issue, so loads in flight cover the WMMA block.
// ---------------------------------------------------------------------------
__global__ void gemm_wmma_kernel(const bf16_t* __restrict__ A,
                                 const bf16_t* __restrict__ Bt,
                                 const float* __restrict__ bias,
                                 const float* __restrict__ residual,
                                 float* __restrict__ outF,
                                 bf16_t* __restrict__ outB,
                                 int N, int K, int mode, float scale) {
  int wave = threadIdx.x >> 5;
  int lane = threadIdx.x & 31;
  int rlow = lane & 15;        // A row within tile / B col within tile
  int hi   = lane >> 4;        // half-wave select

  int m0 = blockIdx.x * 128 + wave * 16;
  int n0 = blockIdx.y * 64;

  v8f acc[4] = {};

  const bf16_t* Arow = A + (size_t)(m0 + rlow) * K + hi * 16;
  const bf16_t* Brow[4];
#pragma unroll
  for (int j = 0; j < 4; ++j)
    Brow[j] = Bt + (size_t)(n0 + j * 16 + rlow) * K + hi * 16;

  // ---- prologue: fragments for kk = 0 ----
  v16bf af = load_afrag(Arow);
  v16bf bf[4];
#pragma unroll
  for (int j = 0; j < 4; ++j) bf[j] = load_bfrag(Brow[j]);

  // ---- steady state: prefetch kk+32 while multiplying kk ----
  for (int kk = 0; kk < K - 32; kk += 32) {
    v16bf afn = load_afrag(Arow + kk + 32);
    v16bf bfn[4];
#pragma unroll
    for (int j = 0; j < 4; ++j) bfn[j] = load_bfrag(Brow[j] + kk + 32);
#pragma unroll
    for (int j = 0; j < 4; ++j) acc[j] = wmma_bf16(af, bf[j], acc[j]);
    af = afn;
#pragma unroll
    for (int j = 0; j < 4; ++j) bf[j] = bfn[j];
  }
  // ---- epilogue step ----
#pragma unroll
  for (int j = 0; j < 4; ++j) acc[j] = wmma_bf16(af, bf[j], acc[j]);

  // Epilogue. C layout: lane holds col n = rlow, rows m = hi*8 + r (r = 0..7).
#pragma unroll
  for (int j = 0; j < 4; ++j) {
    int n = n0 + j * 16 + rlow;
    float bval = bias[n];
#pragma unroll
    for (int r = 0; r < 8; ++r) {
      int m = m0 + hi * 8 + r;
      float val = acc[j][r] + bval;
      if (mode == 0) {                 // Q/K: [B,H,S,Dk], optional 1/sqrt(Dk)
        val *= scale;
        int h = n >> 6, d = n & 63;
        int b = m >> 11, s = m & (SEQ - 1);
        outB[(((size_t)(b * NHEAD + h) * SEQ + s) << 6) + d] = (bf16_t)val;
      } else if (mode == 1) {          // V transposed: [B,H,Dk,S]
        int h = n >> 6, d = n & 63;
        int b = m >> 11, s = m & (SEQ - 1);
        outB[((size_t)(b * NHEAD + h) * DKEY + d) * SEQ + s] = (bf16_t)val;
      } else if (mode == 2) {          // f32 + residual
        size_t o = (size_t)m * N + n;
        outF[o] = val + residual[o];
      } else {                         // relu -> bf16
        outB[(size_t)m * N + n] = (bf16_t)fmaxf(val, 0.0f);
      }
    }
  }
}

// ---------------------------------------------------------------------------
// Flash attention (streaming softmax, never materializes 2048x2048 scores).
// Q pre-scaled by 1/8. Grid: (S/128, B*H); block 256 = 8 waves; each wave
// owns a 16-query tile and streams 32 keys/iteration:
//   K and V fragments are loaded together up front (V does not depend on the
//   scores), so V loads overlap the softmax VALU/transcendental chain and the
//   LDS P-bounce instead of stalling the P@V WMMAs.
// ---------------------------------------------------------------------------
__global__ void flash_attn_kernel(const bf16_t* __restrict__ q,
                                  const bf16_t* __restrict__ kmat,
                                  const bf16_t* __restrict__ vt,
                                  const int* __restrict__ mask,
                                  bf16_t* __restrict__ ctx) {
  int wave = threadIdx.x >> 5;
  int lane = threadIdx.x & 31;
  int rlow = lane & 15;
  int hi   = lane >> 4;

  int bh = blockIdx.y;
  int b  = bh >> 4;
  int h  = bh & 15;
  int q0 = blockIdx.x * 128 + wave * 16;

  __shared__ bf16_t psh[8][16 * 32];     // per-wave P tile (16 q x 32 keys)
  bf16_t* pw = psh[wave];

  // Q fragments (K' = 0..31 and 32..63 of Dk=64), A layout
  const bf16_t* qp = q + ((size_t)bh * SEQ + q0 + rlow) * DKEY + hi * 16;
  v16bf qf0 = load_afrag(qp);
  v16bf qf1 = load_afrag(qp + 32);

  float rmax[8], rsum[8];
#pragma unroll
  for (int r = 0; r < 8; ++r) { rmax[r] = -3.0e38f; rsum[r] = 0.0f; }
  v8f o[4] = {};

  const size_t kbase = (size_t)bh * SEQ * DKEY;
  const size_t vbase = (size_t)bh * DKEY * SEQ;
  const size_t mbase = (size_t)b * SEQ * SEQ;

  for (int kb = 0; kb < SEQ; kb += 32) {
    // ---- issue all global loads for this step: 4 K frags + 4 V frags ----
    const bf16_t* kp0 = kmat + kbase + (size_t)(kb + rlow) * DKEY + hi * 16;
    const bf16_t* kp1 = kmat + kbase + (size_t)(kb + 16 + rlow) * DKEY + hi * 16;
    v16bf kf00 = load_bfrag(kp0);
    v16bf kf01 = load_bfrag(kp0 + 32);
    v16bf kf10 = load_bfrag(kp1);
    v16bf kf11 = load_bfrag(kp1 + 32);
    v16bf vf[4];
#pragma unroll
    for (int j = 0; j < 4; ++j) {
      const bf16_t* vp = vt + vbase + (size_t)(j * 16 + rlow) * SEQ + kb + hi * 16;
      vf[j] = load_bfrag(vp);
    }

    // ---- scores: two 16x16 key tiles ----
    v8f s0 = {}, s1 = {};
    s0 = wmma_bf16(qf0, kf00, s0);
    s0 = wmma_bf16(qf1, kf01, s0);
    s1 = wmma_bf16(qf0, kf10, s1);
    s1 = wmma_bf16(qf1, kf11, s1);

    // ---- mask + online softmax (per q row, 16-lane reductions) ----
    int key_a = kb + rlow;
    int key_b = kb + 16 + rlow;
#pragma unroll
    for (int r = 0; r < 8; ++r) {
      int qrow = q0 + hi * 8 + r;
      const int* mrow = mask + mbase + (size_t)qrow * SEQ;
      float a = s0[r], c = s1[r];
      if (mrow[key_a] == 0) a = -1.0e9f;
      if (mrow[key_b] == 0) c = -1.0e9f;
      float mx = fmaxf(a, c);
#pragma unroll
      for (int off = 8; off > 0; off >>= 1) mx = fmaxf(mx, __shfl_xor(mx, off, 16));
      float mnew = fmaxf(rmax[r], mx);
      float corr = __expf(rmax[r] - mnew);
      rmax[r] = mnew;
      float p0 = __expf(a - mnew);
      float p1 = __expf(c - mnew);
      float ps = p0 + p1;
#pragma unroll
      for (int off = 8; off > 0; off >>= 1) ps += __shfl_xor(ps, off, 16);
      rsum[r] = rsum[r] * corr + ps;
      o[0][r] *= corr; o[1][r] *= corr; o[2][r] *= corr; o[3][r] *= corr;
      // stash P in C-layout
      pw[(hi * 8 + r) * 32 + rlow]      = (bf16_t)p0;
      pw[(hi * 8 + r) * 32 + 16 + rlow] = (bf16_t)p1;
    }
    __syncthreads();

    // ---- P fragment in A layout from LDS ----
    v16bf pf = load_afrag(pw + rlow * 32 + hi * 16);

    // ---- O += P @ V  (V fragments already resident) ----
#pragma unroll
    for (int j = 0; j < 4; ++j) o[j] = wmma_bf16(pf, vf[j], o[j]);
    __syncthreads();
  }

  // ---- normalize + write ctx as [B,S,D] with heads concatenated ----
#pragma unroll
  for (int j = 0; j < 4; ++j) {
#pragma unroll
    for (int r = 0; r < 8; ++r) {
      int qrow = q0 + hi * 8 + r;
      float val = o[j][r] / rsum[r];
      ctx[((size_t)b * SEQ + qrow) * DMODEL + h * DKEY + j * 16 + rlow] = (bf16_t)val;
    }
  }
}

// ---------------------------------------------------------------------------
// Host-side orchestration
// ---------------------------------------------------------------------------
extern "C" void kernel_launch(void* const* d_in, const int* in_sizes, int n_in,
                              void* d_out, int out_size, void* d_ws, size_t ws_size,
                              hipStream_t stream) {
  (void)in_sizes; (void)n_in; (void)out_size; (void)ws_size;

  const float* x    = (const float*)d_in[0];
  const int*   mask = (const int*)d_in[1];
  const float* wq = (const float*)d_in[2];   const float* bq = (const float*)d_in[3];
  const float* wk = (const float*)d_in[4];   const float* bk = (const float*)d_in[5];
  const float* wv = (const float*)d_in[6];   const float* bv = (const float*)d_in[7];
  const float* wo = (const float*)d_in[8];   const float* bo = (const float*)d_in[9];
  const float* w1 = (const float*)d_in[10];  const float* b1 = (const float*)d_in[11];
  const float* w2 = (const float*)d_in[12];  const float* b2 = (const float*)d_in[13];
  const float* ln1a = (const float*)d_in[14]; const float* ln1b = (const float*)d_in[15];
  const float* ln2a = (const float*)d_in[16]; const float* ln2b = (const float*)d_in[17];
  float* out = (float*)d_out;

  // Workspace arena (peak 80 MB with reuse)
  char* ws = (char*)d_ws;
  auto MB = [](size_t m) { return m << 20; };
  bf16_t* wqt = (bf16_t*)(ws + MB(0));    // 2 MB
  bf16_t* wkt = (bf16_t*)(ws + MB(2));    // 2 MB
  bf16_t* wvt = (bf16_t*)(ws + MB(4));    // 2 MB
  bf16_t* wot = (bf16_t*)(ws + MB(6));    // 2 MB
  bf16_t* w1t = (bf16_t*)(ws + MB(8));    // 8 MB
  bf16_t* w2t = (bf16_t*)(ws + MB(16));   // 8 MB
  bf16_t* hn1 = (bf16_t*)(ws + MB(24));   // 8 MB (reused as hn2)
  bf16_t* qb  = (bf16_t*)(ws + MB(32));   // 8 MB
  bf16_t* kb  = (bf16_t*)(ws + MB(40));   // 8 MB
  bf16_t* vtb = (bf16_t*)(ws + MB(48));   // 8 MB
  bf16_t* ctxb= (bf16_t*)(ws + MB(56));   // 8 MB
  float*  x2  = (float*)(ws + MB(64));    // 16 MB
  bf16_t* hn2 = hn1;                      // reuse (hn1 dead after QKV)
  bf16_t* a1  = qb;                       // reuse q/k/vt/ctx 32 MB (dead after x2)

  // 1) weights -> bf16, transposed [N][K]
  transpose_to_bf16_kernel<<<(DMODEL * DMODEL) / 256, 256, 0, stream>>>(wq, wqt, DMODEL, DMODEL);
  transpose_to_bf16_kernel<<<(DMODEL * DMODEL) / 256, 256, 0, stream>>>(wk, wkt, DMODEL, DMODEL);
  transpose_to_bf16_kernel<<<(DMODEL * DMODEL) / 256, 256, 0, stream>>>(wv, wvt, DMODEL, DMODEL);
  transpose_to_bf16_kernel<<<(DMODEL * DMODEL) / 256, 256, 0, stream>>>(wo, wot, DMODEL, DMODEL);
  transpose_to_bf16_kernel<<<(DMODEL * DFF) / 256, 256, 0, stream>>>(w1, w1t, DMODEL, DFF);
  transpose_to_bf16_kernel<<<(DFF * DMODEL) / 256, 256, 0, stream>>>(w2, w2t, DFF, DMODEL);

  // 2) LN1
  layernorm_bf16_kernel<<<TOK, 256, 0, stream>>>(x, hn1, ln1a, ln1b);

  // 3) Q,K,V projections (Q scaled by 1/sqrt(Dk); V stored transposed)
  dim3 gD(TOK / 128, DMODEL / 64);
  gemm_wmma_kernel<<<gD, 256, 0, stream>>>(hn1, wqt, bq, nullptr, nullptr, qb,
                                           DMODEL, DMODEL, 0, 0.125f);
  gemm_wmma_kernel<<<gD, 256, 0, stream>>>(hn1, wkt, bk, nullptr, nullptr, kb,
                                           DMODEL, DMODEL, 0, 1.0f);
  gemm_wmma_kernel<<<gD, 256, 0, stream>>>(hn1, wvt, bv, nullptr, nullptr, vtb,
                                           DMODEL, DMODEL, 1, 1.0f);

  // 4) flash attention
  dim3 gA(SEQ / 128, BATCH * NHEAD);
  flash_attn_kernel<<<gA, 256, 0, stream>>>(qb, kb, vtb, mask, ctxb);

  // 5) output projection + residual -> x2 (fp32)
  gemm_wmma_kernel<<<gD, 256, 0, stream>>>(ctxb, wot, bo, x, x2, nullptr,
                                           DMODEL, DMODEL, 2, 1.0f);

  // 6) LN2
  layernorm_bf16_kernel<<<TOK, 256, 0, stream>>>(x2, hn2, ln2a, ln2b);

  // 7) FFN1: relu(hn2 @ w1 + b1) -> a1 (bf16)
  dim3 gF(TOK / 128, DFF / 64);
  gemm_wmma_kernel<<<gF, 256, 0, stream>>>(hn2, w1t, b1, nullptr, nullptr, a1,
                                           DFF, DMODEL, 3, 1.0f);

  // 8) FFN2 + residual -> out (fp32)
  gemm_wmma_kernel<<<gD, 256, 0, stream>>>(a1, w2t, b2, x2, out, nullptr,
                                           DMODEL, DFF, 2, 1.0f);
}
